// MultiHeadAttention_29678224015664
// MI455X (gfx1250) — matrix operand order repaired and checked by reading, not compiled, not run on hardware
//
#include <hip/hip_runtime.h>
#include <hip/hip_bf16.h>

// ---------------------------------------------------------------------------
// MHA forward for B=2, S=2048, D=1024, H=16, DK=64 on gfx1250 (CDNA5).
// All matmuls via v_wmma_f32_16x16x32_f16 (f16 operands, fp32 accumulate).
//
// Fragment layouts per CDNA5 ISA §7.12.2/.4:
//   A (16x32 f16): lane m = lane&15; VGPR0-3 hold K = kb..kb+7,
//                  VGPR4-7 hold K = kb+16..kb+23, kb = (lane>=16)?8:0.
//   B (32x16 f16): lane n = lane&15; VGPR0-3 hold K = kb..kb+7,
//                  VGPR4-7 hold K = kb+16..kb+23 (same as A).
//   C/D (16x16 f32): lane n = lane&15; VGPR r -> row r + 8*(lane>=16).
// ---------------------------------------------------------------------------

typedef _Float16 v16h __attribute__((ext_vector_type(16)));
typedef _Float16 h8   __attribute__((ext_vector_type(8)));
typedef _Float16 h4   __attribute__((ext_vector_type(4)));
typedef float    v8f  __attribute__((ext_vector_type(8)));

union Frag16 { v16h v; h8 h[2]; };

constexpr int Bc  = 2;
constexpr int Sc  = 2048;
constexpr int Dc  = 1024;
constexpr int Hc  = 16;
constexpr int DKc = 64;

#define GEMM_BM 128
#define GEMM_BN 128
#define GEMM_BK 32

enum { OUT_F32_RM = 0, OUT_F16_HEADS = 1, OUT_F16_HEADS_T = 2 };

__device__ __forceinline__ v8f wmma_f16(const Frag16& a, const Frag16& b, v8f c) {
  return __builtin_amdgcn_wmma_f32_16x16x32_f16(
      /*neg_a=*/false, a.v, /*neg_b=*/false, b.v,
      /*c_mod=*/(short)0, c, /*reuse_a=*/false, /*reuse_b=*/false);
}

// out = (A[M,K] @ W[N,K]^T + bias[N]) * scale, f16 operands staged via LDS.
template <int OUT_MODE>
__global__ __launch_bounds__(256)
void gemm_xwT(const float* __restrict__ A, const float* __restrict__ W,
              const float* __restrict__ bias, void* __restrict__ Out,
              int M, int N, int K, float scale) {
  __shared__ _Float16 As[GEMM_BM][GEMM_BK + 8];
  __shared__ _Float16 Ws[GEMM_BN][GEMM_BK + 8];

  const int tid   = threadIdx.x;
  const int lane  = tid & 31;
  const int wave  = tid >> 5;       // 0..7
  const int wm    = wave & 3;       // wave row (4 x 32 = 128)
  const int wn    = wave >> 2;      // wave col (2 x 64 = 128)
  const int l15   = lane & 15;
  const int lhalf = lane >> 4;
  const int kbA   = lhalf * 8;      // A frag K base (split pattern)
  const int kbB   = lhalf * 16;     // contiguous-pattern K base, unused

  const int bm0 = blockIdx.x * GEMM_BM;
  const int bn0 = blockIdx.y * GEMM_BN;

  v8f acc[2][4] = {};

  for (int k0 = 0; k0 < K; k0 += GEMM_BK) {
    // ---- stage 128x32 A and W tiles, fp32 -> f16 ----
    #pragma unroll
    for (int i = 0; i < 4; ++i) {
      const int e    = i * 256 + tid;     // float4 chunk id, 1024 total
      const int row  = e >> 3;            // 8 chunks per row
      const int col4 = (e & 7) * 4;
      const float4 a4 = *(const float4*)&A[(size_t)(bm0 + row) * K + k0 + col4];
      h4 ah; ah.x = (_Float16)a4.x; ah.y = (_Float16)a4.y;
             ah.z = (_Float16)a4.z; ah.w = (_Float16)a4.w;
      *(h4*)&As[row][col4] = ah;
      const float4 w4 = *(const float4*)&W[(size_t)(bn0 + row) * K + k0 + col4];
      h4 wh; wh.x = (_Float16)w4.x; wh.y = (_Float16)w4.y;
             wh.z = (_Float16)w4.z; wh.w = (_Float16)w4.w;
      *(h4*)&Ws[row][col4] = wh;
      if (k0 + GEMM_BK < K) {  // prefetch next tiles (global_prefetch_b8)
        __builtin_prefetch(&A[(size_t)(bm0 + row) * K + k0 + GEMM_BK + col4], 0, 1);
        __builtin_prefetch(&W[(size_t)(bn0 + row) * K + k0 + GEMM_BK + col4], 0, 1);
      }
    }
    __syncthreads();

    Frag16 af[2], bf[4];
    #pragma unroll
    for (int i = 0; i < 2; ++i) {
      const int r = wm * 32 + i * 16 + l15;
      af[i].h[0] = *(const h8*)&As[r][kbA];
      af[i].h[1] = *(const h8*)&As[r][kbA + 16];
    }
    #pragma unroll
    for (int j = 0; j < 4; ++j) {
      const int c = wn * 64 + j * 16 + l15;
      bf[j].h[0] = *(const h8*)&Ws[c][kbA];
      bf[j].h[1] = *(const h8*)&Ws[c][kbA + 16];
    }
    #pragma unroll
    for (int i = 0; i < 2; ++i)
      #pragma unroll
      for (int j = 0; j < 4; ++j)
        acc[i][j] = wmma_f16(af[i], bf[j], acc[i][j]);
    __syncthreads();
  }

  // ---- epilogue: bias + scale, layout-specific store ----
  #pragma unroll
  for (int j = 0; j < 4; ++j) {
    const int n  = bn0 + wn * 64 + j * 16 + l15;
    const float bn = bias[n];
    #pragma unroll
    for (int i = 0; i < 2; ++i) {
      #pragma unroll
      for (int r = 0; r < 8; ++r) {
        const int m = bm0 + wm * 32 + i * 16 + r + 8 * lhalf;
        const float val = (acc[i][j][r] + bn) * scale;
        if constexpr (OUT_MODE == OUT_F32_RM) {
          ((float*)Out)[(size_t)m * N + n] = val;
        } else {
          const int b  = m / Sc, s = m % Sc;
          const int h  = n / DKc, dk = n % DKc;
          if constexpr (OUT_MODE == OUT_F16_HEADS) {
            ((_Float16*)Out)[(((size_t)(b * Hc + h)) * Sc + s) * DKc + dk] =
                (_Float16)val;
          } else {  // OUT_F16_HEADS_T : [B,H,DK,S]
            ((_Float16*)Out)[(((size_t)(b * Hc + h)) * DKc + dk) * Sc + s] =
                (_Float16)val;
          }
        }
      }
    }
  }
}

// ---------------------------------------------------------------------------
// Causal flash attention: grid (S/64, B*H), 128 threads = 4 waves.
// Wave w owns query rows [16w, 16w+16); thread t<64 owns softmax row t.
// Q pre-scaled by 1/sqrt(DK). Vt is [B,H,DK,S] so PV B-frags are contiguous.
// ---------------------------------------------------------------------------
__global__ __launch_bounds__(128)
void attn_kernel(const _Float16* __restrict__ Qh, const _Float16* __restrict__ Kh,
                 const _Float16* __restrict__ Vt, float* __restrict__ Ctx) {
  __shared__ _Float16 Qs[64][72];
  __shared__ _Float16 Ks[64][72];
  __shared__ _Float16 Vts[64][72];   // [dk][key]
  __shared__ float    Ssm[64][68];
  __shared__ _Float16 Ps[64][72];
  __shared__ float    sStat[64];

  const int tid   = threadIdx.x;
  const int lane  = tid & 31;
  const int wave  = tid >> 5;        // 0..3
  const int l15   = lane & 15;
  const int lhalf = lane >> 4;
  const int kbA   = lhalf * 8;       // A frag K base (split pattern)
  const int kbB   = lhalf * 16;      // contiguous-pattern K base, unused

  const int qt = blockIdx.x;
  const int bh = blockIdx.y;
  const int q0 = qt * 64;

  const _Float16* qbase = Qh + ((size_t)bh * Sc + q0) * DKc;
  const _Float16* kbase = Kh + (size_t)bh * Sc * DKc;
  const _Float16* vbase = Vt + (size_t)bh * DKc * Sc;

  // load Q tile (64 x 64 f16)
  #pragma unroll
  for (int i = 0; i < 4; ++i) {
    const int e = i * 128 + tid, row = e >> 3, off = (e & 7) * 8;
    *(h8*)&Qs[row][off] = *(const h8*)&qbase[row * DKc + off];
  }

  float m_i = -1e30f, l_i = 0.f;     // owned by tid < 64
  v8f acc[4] = {};

  for (int kt = 0; kt <= qt; ++kt) {
    const int k0 = kt * 64;
    #pragma unroll
    for (int i = 0; i < 4; ++i) {
      const int e = i * 128 + tid, row = e >> 3, off = (e & 7) * 8;
      *(h8*)&Ks[row][off]  = *(const h8*)&kbase[(k0 + row) * DKc + off];
      *(h8*)&Vts[row][off] = *(const h8*)&vbase[row * Sc + k0 + off];
      if (kt < qt)
        __builtin_prefetch(&kbase[(k0 + 64 + row) * DKc + off], 0, 1);
    }
    __syncthreads();

    // ---- scores: S = Q K^T (Q pre-scaled), wave row-strip x 64 keys ----
    #pragma unroll
    for (int j = 0; j < 4; ++j) {
      v8f sfrag = {};
      #pragma unroll
      for (int kc = 0; kc < 64; kc += 32) {
        Frag16 a, b;
        const int r = wave * 16 + l15;
        a.h[0] = *(const h8*)&Qs[r][kc + kbA];
        a.h[1] = *(const h8*)&Qs[r][kc + kbA + 16];
        const int c = j * 16 + l15;        // key index within tile
        b.h[0] = *(const h8*)&Ks[c][kc + kbA];
        b.h[1] = *(const h8*)&Ks[c][kc + kbA + 16];
        sfrag = wmma_f16(a, b, sfrag);
      }
      const int col = j * 16 + l15;
      #pragma unroll
      for (int r = 0; r < 8; ++r)
        Ssm[wave * 16 + r + 8 * lhalf][col] = sfrag[r];
    }
    __syncthreads();

    // ---- online softmax, one thread per query row ----
    if (tid < 64) {
      const int qg = q0 + tid;
      const int kmax = min(64, qg - k0 + 1);   // causal: cols [0,kmax) valid
      float mnew = m_i;
      for (int j = 0; j < kmax; ++j) mnew = fmaxf(mnew, Ssm[tid][j]);
      const float alpha = __expf(m_i - mnew);
      float rs = 0.f;
      for (int j = 0; j < 64; ++j) {
        const float p = (j < kmax) ? __expf(Ssm[tid][j] - mnew) : 0.f;
        Ps[tid][j] = (_Float16)p;
        rs += p;
      }
      l_i = l_i * alpha + rs;
      m_i = mnew;
      sStat[tid] = alpha;
    }
    __syncthreads();

    // ---- rescale accumulators, then acc += P @ V ----
    float al[8];
    #pragma unroll
    for (int r = 0; r < 8; ++r) al[r] = sStat[wave * 16 + r + 8 * lhalf];
    #pragma unroll
    for (int j = 0; j < 4; ++j)
      #pragma unroll
      for (int r = 0; r < 8; ++r) acc[j][r] *= al[r];

    #pragma unroll
    for (int kc = 0; kc < 64; kc += 32) {
      Frag16 a;
      const int r = wave * 16 + l15;
      a.h[0] = *(const h8*)&Ps[r][kc + kbA];
      a.h[1] = *(const h8*)&Ps[r][kc + kbA + 16];
      #pragma unroll
      for (int j = 0; j < 4; ++j) {
        Frag16 b;
        const int c = j * 16 + l15;          // dk index
        b.h[0] = *(const h8*)&Vts[c][kc + kbA];
        b.h[1] = *(const h8*)&Vts[c][kc + kbA + 16];
        acc[j] = wmma_f16(a, b, acc[j]);
      }
    }
    __syncthreads();
  }

  // ---- normalize and store context fp32 [B,S,D] ----
  if (tid < 64) sStat[tid] = 1.0f / l_i;
  __syncthreads();
  float il[8];
  #pragma unroll
  for (int r = 0; r < 8; ++r) il[r] = sStat[wave * 16 + r + 8 * lhalf];
  const int b = bh >> 4, h = bh & 15;
  #pragma unroll
  for (int j = 0; j < 4; ++j) {
    const int dk = j * 16 + l15;
    #pragma unroll
    for (int r = 0; r < 8; ++r) {
      const int s = q0 + wave * 16 + r + 8 * lhalf;
      Ctx[((size_t)(b * Sc + s)) * Dc + h * DKc + dk] = acc[j][r] * il[r];
    }
  }
}

// ---------------------------------------------------------------------------
extern "C" void kernel_launch(void* const* d_in, const int* in_sizes, int n_in,
                              void* d_out, int out_size, void* d_ws, size_t ws_size,
                              hipStream_t stream) {
  const float* query = (const float*)d_in[0];
  const float* key_  = (const float*)d_in[1];
  const float* value = (const float*)d_in[2];
  // d_in[3] = attn_mask (causal, applied analytically)
  const float* w_q = (const float*)d_in[4];
  const float* b_q = (const float*)d_in[5];
  const float* w_k = (const float*)d_in[6];
  const float* b_k = (const float*)d_in[7];
  const float* w_v = (const float*)d_in[8];
  const float* b_v = (const float*)d_in[9];
  const float* w_o = (const float*)d_in[10];
  const float* b_o = (const float*)d_in[11];
  float* out = (float*)d_out;

  char* ws = (char*)d_ws;
  _Float16* qh  = (_Float16*)(ws);                        //  8 MB [B,H,S,DK]
  _Float16* kh  = (_Float16*)(ws + ((size_t)8  << 20));   //  8 MB [B,H,S,DK]
  _Float16* vt  = (_Float16*)(ws + ((size_t)16 << 20));   //  8 MB [B,H,DK,S]
  float*    ctx = (float*)   (ws + ((size_t)24 << 20));   // 16 MB [B,S,D]

  const int M = Bc * Sc, N = Dc, K = Dc;
  const dim3 gg(M / GEMM_BM, N / GEMM_BN);
  const dim3 bb(256);
  const float qscale = 0.125f;  // 1/sqrt(DK)

  gemm_xwT<OUT_F16_HEADS>  <<<gg, bb, 0, stream>>>(query, w_q, b_q, qh, M, N, K, qscale);
  gemm_xwT<OUT_F16_HEADS>  <<<gg, bb, 0, stream>>>(key_,  w_k, b_k, kh, M, N, K, 1.0f);
  gemm_xwT<OUT_F16_HEADS_T><<<gg, bb, 0, stream>>>(value, w_v, b_v, vt, M, N, K, 1.0f);

  attn_kernel<<<dim3(Sc / 64, Bc * Hc), dim3(128), 0, stream>>>(qh, kh, vt, ctx);

  gemm_xwT<OUT_F32_RM>     <<<gg, bb, 0, stream>>>(ctx, w_o, b_o, out, M, N, K, 1.0f);
}
